// RGCN4_82154134438202
// MI455X (gfx1250) — matrix operand order repaired and checked by reading, not compiled, not run on hardware
//
#include <hip/hip_runtime.h>
#include <hip/hip_bf16.h>

typedef __attribute__((ext_vector_type(16))) __bf16 v16bf;
typedef __attribute__((ext_vector_type(8)))  float  v8f;

constexpr int N_NODES = 50000;
constexpr int N_EDGES = 800000;
constexpr int R_REL   = 3;
constexpr int IN_F    = 128;
constexpr int HID_F   = 128;
constexpr int D_F     = 384;
constexpr int H_HEADS = 8;
constexpr int O_DIM   = 16;
constexpr int OUT_F   = 64;
constexpr int L_LAYERS = 4;
constexpr float BN_EPS = 1e-5f;

__device__ __forceinline__ float atomAddF(float* p, float v) {
  return __hip_atomic_fetch_add(p, v, __ATOMIC_RELAXED, __HIP_MEMORY_SCOPE_AGENT);
}
__device__ __forceinline__ unsigned encodeF(float f) {
  unsigned u = __float_as_uint(f);
  return (u & 0x80000000u) ? ~u : (u | 0x80000000u);
}
__device__ __forceinline__ float decodeF(unsigned e) {
  unsigned b = (e & 0x80000000u) ? (e & 0x7FFFFFFFu) : ~e;
  return __uint_as_float(b);
}

// ---------------------------------------------------------------------------
// Weight swizzle: W[K, Ncols] f32 row-major -> fragment-ordered bf16:
//   Wsw[kc][t][lane][e] = W[kc*32 + (lane<16 ? 0 : 16) + e][t*16 + (lane&15)]
// ---------------------------------------------------------------------------
__global__ __launch_bounds__(256) void wswizzle_kernel(const float* __restrict__ W,
                                                       __bf16* __restrict__ Wsw,
                                                       int K, int Ncols) {
  int idx = blockIdx.x * 256 + threadIdx.x;
  int total = (K / 32) * (Ncols / 16) * 32;
  if (idx >= total) return;
  int lane = idx & 31;
  int rest = idx >> 5;
  int nt = Ncols / 16;
  int t = rest % nt;
  int kc = rest / nt;
  int col = t * 16 + (lane & 15);
  int krow = kc * 32 + ((lane < 16) ? 0 : 16);
  __bf16* dst = Wsw + (size_t)idx * 16;
#pragma unroll
  for (int e = 0; e < 16; e++)
    dst[e] = (__bf16)W[(size_t)(krow + e) * Ncols + col];
}

// ---------------------------------------------------------------------------
// LDS-free WMMA GEMM: Y[M, Ncols] = X[M, K] @ W (pre-swizzled bf16).
// 256 threads = 8 waves; block tile 128 rows x (NT*16) cols; wave w owns 16 rows.
// All NT B fragments are loaded as one clause, then NT WMMAs consume them.
// ---------------------------------------------------------------------------
template <int NT>
__global__ __launch_bounds__(256) void gemm_bf16_kernel(
    const float* __restrict__ X, const __bf16* __restrict__ Wsw,
    float* __restrict__ Y, int Mtot, int K, int Ncols)
{
  const int wave = threadIdx.x >> 5;
  const int lane = threadIdx.x & 31;
  const int rowBase = blockIdx.x * 128;
  const int ntilesTot = Ncols >> 4;
  const int t0 = blockIdx.y * NT;

  const int m = lane & 15;
  const int lo = (lane < 16) ? 0 : 16;
  const int gr = rowBase + wave * 16 + m;
  const int grc = gr < Mtot ? gr : (Mtot - 1);   // clamp: stores are guarded
  const float* xrow = X + (size_t)grc * K;

  v8f acc[NT];
#pragma unroll
  for (int t = 0; t < NT; t++)
#pragma unroll
    for (int q = 0; q < 8; q++) acc[t][q] = 0.f;

  const int KC = K >> 5;
  for (int kc = 0; kc < KC; ++kc) {
    const int k0 = kc << 5;
    if (kc + 1 < KC) __builtin_prefetch(xrow + k0 + 32, 0, 1);  // global_prefetch_b8
    v8f flo = *(const v8f*)(xrow + k0 + lo);
    v8f fhi = *(const v8f*)(xrow + k0 + lo + 8);
    const __bf16* wbase = Wsw + (((size_t)kc * ntilesTot + t0) * 32 + lane) * 16;
    v16bf bfr[NT];
#pragma unroll
    for (int t = 0; t < NT; t++)
      bfr[t] = *(const v16bf*)(wbase + (size_t)t * 32 * 16);
    v16bf a;
#pragma unroll
    for (int e = 0; e < 8; e++) {
      a[e]     = (__bf16)flo[e];
      a[8 + e] = (__bf16)fhi[e];
    }
#pragma unroll
    for (int t = 0; t < NT; t++)
      acc[t] = __builtin_amdgcn_wmma_f32_16x16x32_bf16(
          false, a, false, bfr[t], (short)0, acc[t], false, false);
  }

  const int half = (lane < 16) ? 0 : 8;
  const int col0 = t0 * 16 + (lane & 15);
#pragma unroll
  for (int t = 0; t < NT; t++) {
    int col = col0 + t * 16;
#pragma unroll
    for (int v = 0; v < 8; v++) {
      int row = rowBase + wave * 16 + half + v;
      if (row < Mtot) Y[(size_t)row * Ncols + col] = acc[t][v];
    }
  }
}

// ---------------------------------------------------------------------------
__global__ void zero_f32_kernel(float* p, int n) {
  int i = blockIdx.x * 256 + threadIdx.x;
  if (i < n) p[i] = 0.f;
}
__global__ void zero_i32_kernel(int* p, int n) {
  int i = blockIdx.x * 256 + threadIdx.x;
  if (i < n) p[i] = 0;
}
__global__ void copy_i32_kernel(const int* s, int* d, int n) {
  int i = blockIdx.x * 256 + threadIdx.x;
  if (i < n) d[i] = s[i];
}

// ---------------- CSR-by-target construction (per relation, edges are static)
__global__ void count_kernel(const int* __restrict__ trg, int* __restrict__ cnt, int E) {
  int e = blockIdx.x * 256 + threadIdx.x;
  if (e < E) atomicAdd(&cnt[trg[e]], 1);
}
// block-wise exclusive scan (256 elements/block) + block totals
__global__ __launch_bounds__(256) void scan_phaseA(const int* __restrict__ cnt,
                                                   int* __restrict__ offs,
                                                   int* __restrict__ bsum, int n) {
  __shared__ int sh[256];
  int tid = threadIdx.x;
  int i = blockIdx.x * 256 + tid;
  int v = (i < n) ? cnt[i] : 0;
  sh[tid] = v;
  __syncthreads();
  for (int d = 1; d < 256; d <<= 1) {
    int t = (tid >= d) ? sh[tid - d] : 0;
    __syncthreads();
    sh[tid] += t;
    __syncthreads();
  }
  if (i < n) offs[i] = sh[tid] - v;  // exclusive
  if (tid == 255) bsum[blockIdx.x] = sh[255];
}
__global__ __launch_bounds__(256) void scan_phaseB(int* __restrict__ bsum, int nb) {
  __shared__ int sh[256];
  int tid = threadIdx.x;
  int v = (tid < nb) ? bsum[tid] : 0;
  sh[tid] = v;
  __syncthreads();
  for (int d = 1; d < 256; d <<= 1) {
    int t = (tid >= d) ? sh[tid - d] : 0;
    __syncthreads();
    sh[tid] += t;
    __syncthreads();
  }
  if (tid < nb) bsum[tid] = sh[tid] - v;  // exclusive block offsets
}
__global__ void scan_phaseC(int* __restrict__ offs, const int* __restrict__ bsum,
                            int n, int total) {
  int i = blockIdx.x * 256 + threadIdx.x;
  if (i < n) offs[i] += bsum[blockIdx.x];
  if (i == 0) offs[n] = total;
}
__global__ void place_kernel(const int* __restrict__ src, const int* __restrict__ trg,
                             int* __restrict__ cur, int* __restrict__ srcSorted, int E) {
  int e = blockIdx.x * 256 + threadIdx.x;
  if (e >= E) return;
  int slot = atomicAdd(&cur[trg[e]], 1);
  srcSorted[slot] = src[e];
}

// ---------------- BatchNorm helpers
__global__ __launch_bounds__(128) void stats_kernel(const float* __restrict__ Y,
                                                    float* __restrict__ stats, int Mtot) {
  int col = threadIdx.x;
  int rowsPer = (Mtot + gridDim.x - 1) / gridDim.x;
  int r0 = blockIdx.x * rowsPer;
  int r1 = min(r0 + rowsPer, Mtot);
  float s = 0.f, sq = 0.f;
  for (int r = r0; r < r1; r++) {
    float v = Y[(size_t)r * 128 + col];
    s += v;
    sq += v * v;
  }
  atomAddF(&stats[col], s);
  atomAddF(&stats[128 + col], sq);
}
__global__ void bn_final_kernel(const float* __restrict__ stats, float* __restrict__ mu,
                                float* __restrict__ rsig, float invN) {
  int c = threadIdx.x;
  float m = stats[c] * invN;
  float var = stats[128 + c] * invN - m * m;
  mu[c] = m;
  rsig[c] = rsqrtf(var + BN_EPS);
}
__global__ __launch_bounds__(256) void bn_apply_kernel(const float* __restrict__ in,
                                                       float* __restrict__ outp,
                                                       const float* __restrict__ mu,
                                                       const float* __restrict__ rsig,
                                                       const float* __restrict__ g,
                                                       const float* __restrict__ b, int n) {
  int i = blockIdx.x * 256 + threadIdx.x;
  if (i >= n) return;
  int c = i & 127;
  float v = (in[i] - mu[c]) * rsig[c] * g[c] + b[c];
  outp[i] = v > 0.f ? v : 0.f;
}

// ---------------- attention scalars
__global__ __launch_bounds__(256) void eler_kernel(const float* __restrict__ hW,
                                                   const float* __restrict__ al,
                                                   const float* __restrict__ ar,
                                                   float* __restrict__ el,
                                                   float* __restrict__ er, int Nn) {
  __shared__ float sal[128], sar[128];
  int tid = threadIdx.x;
  if (tid < 128) { sal[tid] = al[tid]; sar[tid] = ar[tid]; }
  __syncthreads();
  int idx = blockIdx.x * 256 + tid;
  if (idx >= Nn * H_HEADS) return;
  int n = idx >> 3, hh = idx & 7;
  const float* row = hW + (size_t)n * HID_F + hh * O_DIM;
  float sl = 0.f, sr = 0.f;
#pragma unroll
  for (int o = 0; o < O_DIM; o++) {
    float v = row[o];
    sl += v * sal[hh * O_DIM + o];
    sr += v * sar[hh * O_DIM + o];
  }
  el[idx] = sl;
  er[idx] = sr;
}

__global__ void init_smax_kernel(unsigned* smax) { *smax = encodeF(-3.0e38f); }

__global__ __launch_bounds__(256) void smax_kernel(const int* __restrict__ src,
                                                   const int* __restrict__ trg,
                                                   const float* __restrict__ el,
                                                   const float* __restrict__ er,
                                                   unsigned* __restrict__ smax, int Etot) {
  __shared__ float red[256];
  int tid = threadIdx.x;
  int e = blockIdx.x * 256 + tid;
  float m = -3.0e38f;
  if (e < Etot) {
    int s_ = src[e], t_ = trg[e];
#pragma unroll
    for (int hh = 0; hh < H_HEADS; hh++) {
      float v = el[s_ * H_HEADS + hh] + er[t_ * H_HEADS + hh];
      v = v > 0.f ? v : 0.2f * v;
      m = fmaxf(m, v);
    }
  }
  red[tid] = m;
  __syncthreads();
  for (int off = 128; off > 0; off >>= 1) {
    if (tid < off) red[tid] = fmaxf(red[tid], red[tid + off]);
    __syncthreads();
  }
  if (tid == 0) atomicMax(smax, encodeF(red[0]));
}

// ---------------- per-target gather: softmax denom + weighted message sum.
// One thread per (node, head); walks its in-edge segment twice; no atomics.
__global__ __launch_bounds__(256) void gat_gather_kernel(
    const int* __restrict__ offs, const int* __restrict__ srcSorted,
    const float* __restrict__ el, const float* __restrict__ er,
    const unsigned* __restrict__ smaxp, const float* __restrict__ hW,
    float* __restrict__ hnew, int colBase, int Nn)
{
  int idx = blockIdx.x * 256 + threadIdx.x;
  if (idx >= Nn * H_HEADS) return;
  int n = idx >> 3, hh = idx & 7;
  float M = decodeF(*smaxp);
  float ern = er[n * H_HEADS + hh];
  int j0 = offs[n], j1 = offs[n + 1];

  float dsum = 0.f;
  for (int j = j0; j < j1; j++) {
    int s_ = srcSorted[j];
    float v = el[s_ * H_HEADS + hh] + ern;
    v = v > 0.f ? v : 0.2f * v;
    dsum += __expf(v - M);
  }
  float inv = 1.f / (dsum + 1e-16f);

  float4 a0 = {0, 0, 0, 0}, a1 = a0, a2 = a0, a3 = a0;
  for (int j = j0; j < j1; j++) {
    int s_ = srcSorted[j];
    float v = el[s_ * H_HEADS + hh] + ern;
    v = v > 0.f ? v : 0.2f * v;
    float alpha = __expf(v - M) * inv;
    const float4* hs = (const float4*)(hW + (size_t)s_ * HID_F + hh * O_DIM);
    float4 q0 = hs[0], q1 = hs[1], q2 = hs[2], q3 = hs[3];
    a0.x += q0.x * alpha; a0.y += q0.y * alpha; a0.z += q0.z * alpha; a0.w += q0.w * alpha;
    a1.x += q1.x * alpha; a1.y += q1.y * alpha; a1.z += q1.z * alpha; a1.w += q1.w * alpha;
    a2.x += q2.x * alpha; a2.y += q2.y * alpha; a2.z += q2.z * alpha; a2.w += q2.w * alpha;
    a3.x += q3.x * alpha; a3.y += q3.y * alpha; a3.z += q3.z * alpha; a3.w += q3.w * alpha;
  }
  float4* dst = (float4*)(hnew + (size_t)n * D_F + colBase + hh * O_DIM);
  dst[0] = a0; dst[1] = a1; dst[2] = a2; dst[3] = a3;
}

// h = leaky_relu(hnew, 0.01) + x
__global__ __launch_bounds__(256) void combine_kernel(const float* __restrict__ hnew,
                                                      const float* __restrict__ x,
                                                      float* __restrict__ h, int n) {
  int i = blockIdx.x * 256 + threadIdx.x;
  if (i >= n) return;
  float v = hnew[i];
  v = v > 0.f ? v : 0.01f * v;
  h[i] = v + x[i];
}

// ---------------------------------------------------------------------------
extern "C" void kernel_launch(void* const* d_in, const int* in_sizes, int n_in,
                              void* d_out, int out_size, void* d_ws, size_t ws_size,
                              hipStream_t stream) {
  const float* inputs = (const float*)d_in[0];
  const int*   edges  = (const int*)d_in[1];
  const float* W_emb1 = (const float*)d_in[2];
  const float* W_emb2 = (const float*)d_in[3];
  const float* g_emb  = (const float*)d_in[4];
  const float* b_emb  = (const float*)d_in[5];
  const float* Wg     = (const float*)d_in[6];
  const float* al     = (const float*)d_in[7];
  const float* ar     = (const float*)d_in[8];
  const float* W_d1   = (const float*)d_in[9];
  const float* W_d2   = (const float*)d_in[10];
  const float* g_d    = (const float*)d_in[11];
  const float* b_d    = (const float*)d_in[12];
  float* out = (float*)d_out;

  float* ws = (float*)d_ws;
  size_t off = 0;
  float* x     = ws + off; off += (size_t)N_NODES * D_F;
  float* h     = ws + off; off += (size_t)N_NODES * D_F;
  float* hnew  = ws + off; off += (size_t)N_NODES * D_F;
  float* hW    = ws + off; off += (size_t)N_NODES * HID_F;  // h1 / decoder t
  float* tbn   = ws + off; off += (size_t)N_NODES * HID_F;  // relu(bn(.))
  float* el    = ws + off; off += (size_t)N_NODES * H_HEADS;
  float* er    = ws + off; off += (size_t)N_NODES * H_HEADS;
  float* stats = ws + off; off += 256;
  float* mu    = ws + off; off += 128;
  float* rsig  = ws + off; off += 128;
  unsigned* smax = (unsigned*)(ws + off); off += 16;
  // swizzled bf16 weights
  __bf16* swbase = (__bf16*)(ws + off);
  size_t boff = 0;
  __bf16* sw_emb1 = swbase + boff; boff += (size_t)IN_F * HID_F;
  __bf16* sw_emb2 = swbase + boff; boff += (size_t)HID_F * D_F;
  __bf16* sw_g    = swbase + boff; boff += (size_t)L_LAYERS * R_REL * D_F * HID_F;
  __bf16* sw_d1   = swbase + boff; boff += (size_t)D_F * HID_F;
  __bf16* sw_d2   = swbase + boff; boff += (size_t)HID_F * OUT_F;
  off += (boff + 1) / 2;
  // CSR-by-target (per relation)
  int* ibase = (int*)(ws + off);
  size_t ioff = 0;
  int* offs3 = ibase + ioff; ioff += (size_t)R_REL * (N_NODES + 64);
  int* ssrt3 = ibase + ioff; ioff += (size_t)R_REL * N_EDGES;
  int* cnt   = ibase + ioff; ioff += N_NODES;
  int* cur   = ibase + ioff; ioff += N_NODES;
  int* bsum  = ibase + ioff; ioff += 256;

  const dim3 blk(256);
  const int rowBlocks = (N_NODES + 127) / 128;
  const int nhBlocks  = (N_NODES * H_HEADS + 255) / 256;
  const int nHidBlks  = (N_NODES * HID_F + 255) / 256;
  const int eBlocks   = (N_EDGES + 255) / 256;
  const int ndBlocks  = (N_NODES * D_F + 255) / 256;
  const int nBlocks   = (N_NODES + 255) / 256;

  // ---- Build per-relation CSR keyed by target (edges are layer-invariant)
  for (int r = 0; r < R_REL; r++) {
    const int* src = edges + (size_t)r * 2 * N_EDGES;
    const int* trg = src + N_EDGES;
    int* offs = offs3 + (size_t)r * (N_NODES + 64);
    int* ssrt = ssrt3 + (size_t)r * N_EDGES;
    zero_i32_kernel<<<nBlocks, blk, 0, stream>>>(cnt, N_NODES);
    count_kernel<<<eBlocks, blk, 0, stream>>>(trg, cnt, N_EDGES);
    scan_phaseA<<<nBlocks, blk, 0, stream>>>(cnt, offs, bsum, N_NODES);
    scan_phaseB<<<1, blk, 0, stream>>>(bsum, nBlocks);
    scan_phaseC<<<nBlocks, blk, 0, stream>>>(offs, bsum, N_NODES, N_EDGES);
    copy_i32_kernel<<<nBlocks, blk, 0, stream>>>(offs, cur, N_NODES);
    place_kernel<<<eBlocks, blk, 0, stream>>>(src, trg, cur, ssrt, N_EDGES);
  }

  // ---- Pre-swizzle all weights into WMMA fragment order (bf16)
  wswizzle_kernel<<<(IN_F / 32 * HID_F / 16 * 32 + 255) / 256, blk, 0, stream>>>(
      W_emb1, sw_emb1, IN_F, HID_F);
  wswizzle_kernel<<<(HID_F / 32 * D_F / 16 * 32 + 255) / 256, blk, 0, stream>>>(
      W_emb2, sw_emb2, HID_F, D_F);
  for (int i = 0; i < L_LAYERS * R_REL; i++) {
    wswizzle_kernel<<<(D_F / 32 * HID_F / 16 * 32 + 255) / 256, blk, 0, stream>>>(
        Wg + (size_t)i * D_F * HID_F, sw_g + (size_t)i * D_F * HID_F, D_F, HID_F);
  }
  wswizzle_kernel<<<(D_F / 32 * HID_F / 16 * 32 + 255) / 256, blk, 0, stream>>>(
      W_d1, sw_d1, D_F, HID_F);
  wswizzle_kernel<<<(HID_F / 32 * OUT_F / 16 * 32 + 255) / 256, blk, 0, stream>>>(
      W_d2, sw_d2, HID_F, OUT_F);

  // ---- Embedding MLP
  gemm_bf16_kernel<8><<<dim3(rowBlocks, 1), blk, 0, stream>>>(
      inputs, sw_emb1, hW, N_NODES, IN_F, HID_F);
  zero_f32_kernel<<<1, 256, 0, stream>>>(stats, 256);
  stats_kernel<<<98, 128, 0, stream>>>(hW, stats, N_NODES);
  bn_final_kernel<<<1, 128, 0, stream>>>(stats, mu, rsig, 1.0f / N_NODES);
  bn_apply_kernel<<<nHidBlks, blk, 0, stream>>>(hW, tbn, mu, rsig, g_emb, b_emb,
                                                N_NODES * HID_F);
  gemm_bf16_kernel<8><<<dim3(rowBlocks, D_F / 128), blk, 0, stream>>>(
      tbn, sw_emb2, x, N_NODES, HID_F, D_F);

  hipMemcpyAsync(h, x, (size_t)N_NODES * D_F * sizeof(float),
                 hipMemcpyDeviceToDevice, stream);

  // ---- GAT layers
  for (int l = 0; l < L_LAYERS; l++) {
    for (int r = 0; r < R_REL; r++) {
      const __bf16* Wlr = sw_g + (size_t)(l * R_REL + r) * D_F * HID_F;
      const float* allr = al + (size_t)(l * R_REL + r) * HID_F;
      const float* arlr = ar + (size_t)(l * R_REL + r) * HID_F;
      const int* src = edges + (size_t)r * 2 * N_EDGES;
      const int* trg = src + N_EDGES;
      const int* offs = offs3 + (size_t)r * (N_NODES + 64);
      const int* ssrt = ssrt3 + (size_t)r * N_EDGES;

      gemm_bf16_kernel<8><<<dim3(rowBlocks, 1), blk, 0, stream>>>(
          h, Wlr, hW, N_NODES, D_F, HID_F);
      eler_kernel<<<nhBlocks, blk, 0, stream>>>(hW, allr, arlr, el, er, N_NODES);
      init_smax_kernel<<<1, 1, 0, stream>>>(smax);
      smax_kernel<<<eBlocks, blk, 0, stream>>>(src, trg, el, er, smax, N_EDGES);
      gat_gather_kernel<<<nhBlocks, blk, 0, stream>>>(offs, ssrt, el, er, smax, hW,
                                                      hnew, r * HID_F, N_NODES);
    }
    combine_kernel<<<ndBlocks, blk, 0, stream>>>(hnew, x, h, N_NODES * D_F);
  }

  // ---- Decoder MLP
  gemm_bf16_kernel<8><<<dim3(rowBlocks, 1), blk, 0, stream>>>(
      h, sw_d1, hW, N_NODES, D_F, HID_F);
  zero_f32_kernel<<<1, 256, 0, stream>>>(stats, 256);
  stats_kernel<<<98, 128, 0, stream>>>(hW, stats, N_NODES);
  bn_final_kernel<<<1, 128, 0, stream>>>(stats, mu, rsig, 1.0f / N_NODES);
  bn_apply_kernel<<<nHidBlks, blk, 0, stream>>>(hW, tbn, mu, rsig, g_d, b_d,
                                                N_NODES * HID_F);
  gemm_bf16_kernel<4><<<dim3(rowBlocks, 1), blk, 0, stream>>>(
      tbn, sw_d2, out, N_NODES, HID_F, OUT_F);
}
